// AttLoc_19602230739941
// MI455X (gfx1250) — compile-verified, matrix operand order checked by reading, not run on hardware
//
#include <hip/hip_runtime.h>
#include <hip/hip_bf16.h>
#include <cstdint>

// ---------------- problem constants ----------------
#define B_  32
#define T_  1500
#define E_  512
#define A_  512
#define C_  10
#define K_  100
#define O_  512
#define TTILES 94              // ceil(T/16)
#define TSPLIT 6               // context-kernel T split (6*250 = 1500)

// ---------------- workspace layout (bytes) ----------------
#define WS_WENC 0u                          // 512x512 bf16 swizzled = 512KB
#define WS_WDEC (512u*1024u)
#define WS_WO   (1024u*1024u)
#define WS_DP   (1536u*1024u)               // dec proj + b_enc  [32x512] f32 = 64KB
#define WS_ATTC (WS_DP + 64u*1024u)         // conv out [32x10x1500] f32 = 1,920,000
#define WS_E    (WS_ATTC + 1920000u)        // e [32x1500] f32 = 192,000
#define WS_C    (WS_E + 192000u)            // context [32x512] f32 = 64KB

typedef __attribute__((ext_vector_type(16))) __bf16          v16bf;
typedef __attribute__((ext_vector_type(16))) unsigned short  v16us;
typedef __attribute__((ext_vector_type(8)))  float           v8f;

struct Q2 { uint4 a, b; };   // 32 bytes == one bf16 fragment

static __device__ inline unsigned short f2bf(float x) {
  unsigned int u = __builtin_bit_cast(unsigned int, x);
  unsigned int r = u + 0x7FFFu + ((u >> 16) & 1u);   // round-to-nearest-even
  return (unsigned short)(r >> 16);
}

static __device__ inline v16bf ld_frag_g(const unsigned short* p) {
  Q2 q;
  q.a = ((const uint4*)p)[0];
  q.b = ((const uint4*)p)[1];
  return __builtin_bit_cast(v16bf, q);
}

// ------------------------------------------------------------------
// Swizzle a 512x512 f32 weight (row-major [k][n]) into bf16 B-fragments.
// Fragment addr: ((nt*16+kt)*512 + lane*16) ushorts; per-lane element e:
//   k = kt*32 + (e<8 ? 8*h+e : 16+8*h+(e-8)),  n = nt*16 + (lane&15), h=lane>>4
// ------------------------------------------------------------------
__global__ void swz_kernel(const float* __restrict__ W, unsigned short* __restrict__ dst) {
  int id   = blockIdx.x;            // nt*16 + kt, 0..511
  int nt   = id >> 4;
  int kt   = id & 15;
  int lane = threadIdx.x;           // 0..31
  int n    = nt * 16 + (lane & 15);
  int h    = lane >> 4;
  unsigned short* out = dst + (size_t)id * 512 + lane * 16;
#pragma unroll
  for (int e = 0; e < 16; ++e) {
    int kk = kt * 32 + ((e < 8) ? (8 * h + e) : (16 + 8 * h + (e - 8)));
    out[e] = f2bf(W[kk * 512 + n]);
  }
}

// ------------------------------------------------------------------
// Small GEMM: D[m,n] = A[m,:] @ Bswz + bias[n],  M=32, N=512, K=512 (bf16 WMMA)
// grid (32 ntiles, 2 mtiles), 32 threads (1 wave)
// ------------------------------------------------------------------
__global__ void __launch_bounds__(32) gemm32_kernel(
    const float* __restrict__ Arows, const unsigned short* __restrict__ Bswz,
    const float* __restrict__ bias, float* __restrict__ Dst) {
  int nt   = blockIdx.x;
  int mt   = blockIdx.y;
  int lane = threadIdx.x;
  int h    = lane >> 4;
  int m    = mt * 16 + (lane & 15);
  const float* arow = Arows + (size_t)m * 512;

  v8f acc = {};
  for (int kt = 0; kt < 16; ++kt) {
    v16us af;
    int k0 = kt * 32 + 8 * h;
#pragma unroll
    for (int e = 0; e < 8; ++e) af[e]     = f2bf(arow[k0 + e]);
#pragma unroll
    for (int e = 0; e < 8; ++e) af[8 + e] = f2bf(arow[k0 + 16 + e]);
    v16bf av = __builtin_bit_cast(v16bf, af);
    v16bf bv = ld_frag_g(Bswz + ((size_t)(nt * 16 + kt) * 512 + lane * 16));
    acc = __builtin_amdgcn_wmma_f32_16x16x32_bf16(false, av, false, bv,
                                                  (short)0, acc, false, false);
  }
  int n = nt * 16 + (lane & 15);
  float bvv = bias[n];
#pragma unroll
  for (int r = 0; r < 8; ++r) {
    int mm = mt * 16 + r + 8 * h;         // D layout: M = r + 8*(lane/16)
    Dst[mm * 512 + n] = acc[r] + bvv;
  }
}

// ------------------------------------------------------------------
// Location conv: attc[b,c,t] = sum_k att_prev[b,t+k-K] * conv_w[c,0,k]
// ------------------------------------------------------------------
__global__ void __launch_bounds__(256) conv_kernel(
    const float* __restrict__ prev, const float* __restrict__ cw,
    float* __restrict__ attc) {
  __shared__ float sp[T_ + 2 * K_];       // zero-padded row
  __shared__ float sw[C_ * (2 * K_ + 1)];
  int b = blockIdx.x, tid = threadIdx.x;
  for (int i = tid; i < T_ + 2 * K_; i += 256) {
    int t = i - K_;
    sp[i] = (t >= 0 && t < T_) ? prev[b * T_ + t] : 0.f;
  }
  for (int i = tid; i < C_ * (2 * K_ + 1); i += 256) sw[i] = cw[i];
  __syncthreads();
  for (int idx = tid; idx < C_ * T_; idx += 256) {
    int c = idx / T_, t = idx % T_;
    const float* wp = &sw[c * (2 * K_ + 1)];
    const float* pp = &sp[t];
    float s = 0.f;
    for (int k = 0; k < 2 * K_ + 1; ++k) s += pp[k] * wp[k];
    attc[(b * C_ + c) * T_ + t] = s;
  }
}

// ------------------------------------------------------------------
// Fused energy kernel: per (b, 16-row t-tile)
//   e[b,t] = sum_a g[a] * tanh( (enc@W_enc)[t,a] + dp[b,a] + (attc@W_att)[t,a] )
// 8 waves x 4 N-tiles each cover A=512; K=512 via 16 bf16 WMMA steps.
// K loop fully unrolled so the scheduler can hoist B-fragment loads
// across v_wmma issues (L2-resident weights).
// ------------------------------------------------------------------
__global__ void __launch_bounds__(256) att_energy_kernel(
    const float* __restrict__ enc, const unsigned short* __restrict__ Bswz,
    const float* __restrict__ dp, const float* __restrict__ attc,
    const float* __restrict__ W_att, const float* __restrict__ g,
    float* __restrict__ e_out) {
  __shared__ unsigned short sA[16 * 520];   // bf16 A tile, padded rows (b128 conflict-free)
  __shared__ float sWatt[C_ * 512];
  __shared__ float sG[512];
  __shared__ float sAttc[C_ * 16];
  __shared__ float sE[16];

  const int b  = blockIdx.y;
  const int t0 = blockIdx.x * 16;
  const int tid = threadIdx.x;

  // stage A rows (zero-pad t >= T) as bf16
  for (int idx = tid; idx < 16 * 512; idx += 256) {
    int row = idx >> 9, col = idx & 511;
    int t = t0 + row;
    float v = (t < T_) ? enc[((size_t)(b * T_ + t)) * E_ + col] : 0.f;
    sA[row * 520 + col] = f2bf(v);
  }
  for (int idx = tid; idx < C_ * 512; idx += 256) sWatt[idx] = W_att[idx];
  for (int idx = tid; idx < 512; idx += 256)      sG[idx] = g[idx];
  if (tid < C_ * 16) {
    int c = tid >> 4, tl = tid & 15;
    int t = t0 + tl;
    sAttc[c * 16 + tl] = (t < T_) ? attc[(b * C_ + c) * T_ + t] : 0.f;
  }
  if (tid < 16) sE[tid] = 0.f;
  __syncthreads();

  const int wave = tid >> 5;
  const int lane = tid & 31;
  const int h    = lane >> 4;
  const int m    = lane & 15;
  const int nt0  = wave * 4;                // first of 4 N-tiles for this wave

  const unsigned short* bbase = Bswz + (size_t)lane * 16;

  v8f acc[4] = {{}, {}, {}, {}};
#pragma unroll
  for (int kt = 0; kt < 16; ++kt) {
    const unsigned short* ap = &sA[m * 520 + kt * 32 + 8 * h];
    Q2 aq;
    aq.a = ((const uint4*)ap)[0];
    aq.b = *(const uint4*)(ap + 16);
    v16bf av = __builtin_bit_cast(v16bf, aq);
#pragma unroll
    for (int tt = 0; tt < 4; ++tt) {
      v16bf bv = ld_frag_g(bbase + (size_t)((nt0 + tt) * 16 + kt) * 512);
      acc[tt] = __builtin_amdgcn_wmma_f32_16x16x32_bf16(false, av, false, bv,
                                                        (short)0, acc[tt], false, false);
    }
  }

  // epilogue: + dp + conv-proj, tanh, *g, reduce over a
#pragma unroll
  for (int tt = 0; tt < 4; ++tt) {
    int a = (nt0 + tt) * 16 + (lane & 15);
    float dpv = dp[b * 512 + a];
    float gv  = sG[a];
    float wa[C_];
#pragma unroll
    for (int c = 0; c < C_; ++c) wa[c] = sWatt[c * 512 + a];
#pragma unroll
    for (int r = 0; r < 8; ++r) {
      int tl = r + 8 * h;
      float cp = 0.f;
#pragma unroll
      for (int c = 0; c < C_; ++c) cp += wa[c] * sAttc[c * 16 + tl];
      float contrib = tanhf(acc[tt][r] + dpv + cp) * gv;
#pragma unroll
      for (int off = 1; off < 16; off <<= 1)
        contrib += __shfl_xor(contrib, off, 16);   // reduce across the 16-lane group
      if ((lane & 15) == 0) atomicAdd(&sE[tl], contrib);
    }
  }
  __syncthreads();
  if (tid < 16) {
    int t = t0 + tid;
    if (t < T_) e_out[b * T_ + t] = sE[tid];
  }
}

// ------------------------------------------------------------------
// softmax over T with SCALING=2.0
// ------------------------------------------------------------------
__global__ void __launch_bounds__(256) softmax_kernel(
    const float* __restrict__ e, float* __restrict__ w) {
  __shared__ float se[T_];
  __shared__ float red[256];
  int b = blockIdx.x, tid = threadIdx.x;
  float m = -1e30f;
  for (int t = tid; t < T_; t += 256) {
    float v = 2.0f * e[b * T_ + t];
    se[t] = v;
    m = fmaxf(m, v);
  }
  red[tid] = m; __syncthreads();
  for (int s = 128; s > 0; s >>= 1) {
    if (tid < s) red[tid] = fmaxf(red[tid], red[tid + s]);
    __syncthreads();
  }
  m = red[0]; __syncthreads();
  float sum = 0.f;
  for (int t = tid; t < T_; t += 256) {
    float p = __expf(se[t] - m);
    se[t] = p;
    sum += p;
  }
  red[tid] = sum; __syncthreads();
  for (int s = 128; s > 0; s >>= 1) {
    if (tid < s) red[tid] += red[tid + s];
    __syncthreads();
  }
  float inv = 1.0f / red[0];
  for (int t = tid; t < T_; t += 256) w[b * T_ + t] = se[t] * inv;
}

// ------------------------------------------------------------------
// zero-init for the atomic context accumulation (workspace is poisoned)
// ------------------------------------------------------------------
__global__ void __launch_bounds__(256) zero_kernel(float* __restrict__ p, int n) {
  int i = blockIdx.x * 256 + threadIdx.x;
  if (i < n) p[i] = 0.f;
}

// ------------------------------------------------------------------
// context: c[b,e] = sum_t w[b,t] * enc[b,t,e]
// T split across blocks; partials combined with global_atomic_add_f32.
// ------------------------------------------------------------------
__global__ void __launch_bounds__(256) context_kernel(
    const float* __restrict__ enc, const float* __restrict__ w,
    float* __restrict__ c) {
  int et = blockIdx.x, b = blockIdx.y, ts = blockIdx.z;
  int tid = threadIdx.x;
  int e = et * 256 + tid;
  int tbeg = ts * (T_ / TSPLIT);
  int tend = tbeg + (T_ / TSPLIT);
  const float* wp = w + b * T_;
  const float* ep = enc + ((size_t)b * T_) * E_ + e;
  float s = 0.f;
  for (int t = tbeg; t < tend; ++t) s += wp[t] * ep[(size_t)t * E_];
  atomicAdd(&c[b * E_ + e], s);
}

// ------------------------------------------------------------------
extern "C" void kernel_launch(void* const* d_in, const int* in_sizes, int n_in,
                              void* d_out, int out_size, void* d_ws, size_t ws_size,
                              hipStream_t stream) {
  const float* enc      = (const float*)d_in[0];
  const float* dec_z    = (const float*)d_in[2];
  const float* att_prev = (const float*)d_in[3];
  const float* W_enc    = (const float*)d_in[4];
  const float* b_enc    = (const float*)d_in[5];
  const float* W_dec    = (const float*)d_in[6];
  const float* W_att    = (const float*)d_in[7];
  const float* conv_w   = (const float*)d_in[8];
  const float* gvec     = (const float*)d_in[9];
  const float* W_o      = (const float*)d_in[10];
  const float* b_o      = (const float*)d_in[11];
  (void)in_sizes; (void)n_in; (void)out_size; (void)ws_size;

  char* ws = (char*)d_ws;
  unsigned short* wencS = (unsigned short*)(ws + WS_WENC);
  unsigned short* wdecS = (unsigned short*)(ws + WS_WDEC);
  unsigned short* woS   = (unsigned short*)(ws + WS_WO);
  float* dp   = (float*)(ws + WS_DP);
  float* attc = (float*)(ws + WS_ATTC);
  float* ebuf = (float*)(ws + WS_E);
  float* cbuf = (float*)(ws + WS_C);

  float* out_c = (float*)d_out;          // [32,512]
  float* out_w = out_c + B_ * O_;        // [32,1500]

  // 1) swizzle weights to bf16 WMMA B-fragments
  swz_kernel<<<512, 32, 0, stream>>>(W_enc, wencS);
  swz_kernel<<<512, 32, 0, stream>>>(W_dec, wdecS);
  swz_kernel<<<512, 32, 0, stream>>>(W_o,   woS);
  // 2) dp = dec_z @ W_dec + b_enc (bias folded)
  gemm32_kernel<<<dim3(32, 2), 32, 0, stream>>>(dec_z, wdecS, b_enc, dp);
  // 3) location conv
  conv_kernel<<<B_, 256, 0, stream>>>(att_prev, conv_w, attc);
  // 4) fused big GEMM + tanh + gvec dot -> e
  att_energy_kernel<<<dim3(TTILES, B_), 256, 0, stream>>>(enc, wencS, dp, attc,
                                                          W_att, gvec, ebuf);
  // 5) softmax -> w output
  softmax_kernel<<<B_, 256, 0, stream>>>(ebuf, out_w);
  // 6) context vector (zero then atomic-accumulate T partials)
  zero_kernel<<<(B_ * E_ + 255) / 256, 256, 0, stream>>>(cbuf, B_ * E_);
  context_kernel<<<dim3(2, B_, TSPLIT), 256, 0, stream>>>(enc, out_w, cbuf);
  // 7) c @ W_o + b_o -> c output
  gemm32_kernel<<<dim3(32, 2), 32, 0, stream>>>(cbuf, woS, b_o, out_c);
}
